// GCN_E_88081189306887
// MI455X (gfx1250) — matrix operand order repaired and checked by reading, not compiled, not run on hardware
//
#include <hip/hip_runtime.h>
#include <cmath>

typedef __attribute__((ext_vector_type(16))) __bf16 v16bf;
typedef __attribute__((ext_vector_type(8)))  float  v8f;

#define LEAKY_SLOPE 0.25f
#define HBINS 4096

static __device__ __forceinline__ __bf16 f2bf(float f) {
  unsigned int u = __float_as_uint(f);
  u += 0x7FFFu + ((u >> 16) & 1u);            // round to nearest even
  unsigned short s = (unsigned short)(u >> 16);
  __bf16 r;
  __builtin_memcpy(&r, &s, 2);
  return r;
}

// ------------------------------------------------------------------ pad/convert
// out[r*ldOut+c] = (r<rows && c<cols) ? bf16(in[r*cols+c]) : 0   over rowsAlloc x ldOut
__global__ void cvt_pad(const float* __restrict__ in, __bf16* __restrict__ out,
                        int rows, int cols, int rowsAlloc, int ldOut) {
  long long total = (long long)rowsAlloc * ldOut;
  long long i = (long long)blockIdx.x * blockDim.x + threadIdx.x;
  long long stride = (long long)gridDim.x * blockDim.x;
  for (; i < total; i += stride) {
    int r = (int)(i / ldOut), c = (int)(i % ldOut);
    float v = (r < rows && c < cols) ? in[(size_t)r * cols + c] : 0.0f;
    out[i] = f2bf(v);
  }
}

// transposing variant: in is [cols, rows] row-major (a weight [K,N]); out is [rows,cols]=[N,K]
__global__ void cvt_pad_t(const float* __restrict__ in, __bf16* __restrict__ out,
                          int rows, int cols, int rowsAlloc, int ldOut) {
  long long total = (long long)rowsAlloc * ldOut;
  long long i = (long long)blockIdx.x * blockDim.x + threadIdx.x;
  long long stride = (long long)gridDim.x * blockDim.x;
  for (; i < total; i += stride) {
    int r = (int)(i / ldOut), c = (int)(i % ldOut);
    float v = (r < rows && c < cols) ? in[(size_t)c * rows + r] : 0.0f;
    out[i] = f2bf(v);
  }
}

__global__ void zero_cols_bf(__bf16* __restrict__ p, int rows, int c0, int c1, int ld) {
  int w = c1 - c0;
  long long total = (long long)rows * w;
  long long i = (long long)blockIdx.x * blockDim.x + threadIdx.x;
  long long stride = (long long)gridDim.x * blockDim.x;
  unsigned short* q = (unsigned short*)p;
  for (; i < total; i += stride) {
    int r = (int)(i / w), c = c0 + (int)(i % w);
    q[(size_t)r * ld + c] = 0;
  }
}

// ------------------------------------------------------------------ WMMA GEMM
// C[M,N] = A[M,K] @ B^T where B is stored [N,K] row-major (both bf16, K%32==0,
// rows padded so every async tile fetch is in-bounds; K-pad columns are zero).
#define BM 128
#define BN 64
#define BKK 32
#define LDT 40   // LDS pitch (bf16): 80B rows -> every 16B chunk naturally aligned

union Frag { uint4 u[2]; v16bf v; };

__launch_bounds__(256, 2)
__global__ void gemm_bf16_wmma(const __bf16* __restrict__ A, int lda,
                               const __bf16* __restrict__ B, int ldb,
                               int M, int N, int K,
                               const float* __restrict__ bias, int leaky,
                               float* __restrict__ outF,
                               __bf16* __restrict__ outBF, int ldBF, int bfTrans) {
  __shared__ __align__(16) __bf16 As[2][BM * LDT];
  __shared__ __align__(16) __bf16 Bs[2][BN * LDT];

  const int tid  = threadIdx.x;
  const int lane = tid & 31;
  const int wave = tid >> 5;
  const int lm   = lane & 15;
  const int half = lane >> 4;
  const int wmB  = (wave >> 1) << 5;   // 0,32,64,96
  const int wnB  = (wave & 1) << 5;    // 0,32

  const int blockN = blockIdx.x * BN;
  const int blockM = blockIdx.y * BM;

  // LDS byte offsets = low 32 bits of the flat address (ISA 10.2 LDS aperture)
  const unsigned aBase = (unsigned)(size_t)(void*)&As[0][0];
  const unsigned bBase = (unsigned)(size_t)(void*)&Bs[0][0];

  // async tile fill: A = 512 16B chunks (2/thread), B = 256 (1/thread); no guards.
  auto load_stage = [&](int s, int k0) {
    #pragma unroll
    for (int i = 0; i < 2; ++i) {
      int ch = tid + (i << 8);
      int r = ch >> 2, c = (ch & 3) << 3;
      unsigned loff = aBase + (unsigned)(s * (BM * LDT * 2) + r * (LDT * 2) + c * 2);
      const __bf16* gp = A + (size_t)(blockM + r) * lda + (k0 + c);
      asm volatile("global_load_async_to_lds_b128 %0, %1, off"
                   :: "v"(loff), "v"(gp) : "memory");
    }
    {
      int r = tid >> 2, c = (tid & 3) << 3;
      unsigned loff = bBase + (unsigned)(s * (BN * LDT * 2) + r * (LDT * 2) + c * 2);
      const __bf16* gp = B + (size_t)(blockN + r) * ldb + (k0 + c);
      asm volatile("global_load_async_to_lds_b128 %0, %1, off"
                   :: "v"(loff), "v"(gp) : "memory");
    }
  };

  v8f acc[2][2] = {};
  const int nsteps = K >> 5;

  load_stage(0, 0);
  asm volatile("s_wait_asynccnt 0x0" ::: "memory");
  __syncthreads();

  for (int it = 0; it < nsteps; ++it) {
    const int cur = it & 1;
    if (it + 1 < nsteps) load_stage(cur ^ 1, (it + 1) << 5);   // DMA overlaps WMMA

    const __bf16* as = &As[cur][0];
    const __bf16* bs = &Bs[cur][0];
    Frag fa[2], fb[2];
    #pragma unroll
    for (int i = 0; i < 2; ++i) {
      const __bf16* p = as + (wmB + (i << 4) + lm) * LDT;
      fa[i].u[0] = *(const uint4*)(p + (half << 3));
      fa[i].u[1] = *(const uint4*)(p + 16 + (half << 3));
    }
    #pragma unroll
    for (int j = 0; j < 2; ++j) {
      const __bf16* p = bs + (wnB + (j << 4) + lm) * LDT;
      fb[j].u[0] = *(const uint4*)(p + (half << 3));
      fb[j].u[1] = *(const uint4*)(p + 16 + (half << 3));
    }
    #pragma unroll
    for (int i = 0; i < 2; ++i)
      #pragma unroll
      for (int j = 0; j < 2; ++j)
        acc[i][j] = __builtin_amdgcn_wmma_f32_16x16x32_bf16(
            false, fa[i].v, false, fb[j].v, (short)0, acc[i][j], false, false);

    asm volatile("s_wait_asynccnt 0x0" ::: "memory");
    __syncthreads();
  }

  #pragma unroll
  for (int i = 0; i < 2; ++i) {
    #pragma unroll
    for (int j = 0; j < 2; ++j) {
      int gn = blockN + wnB + (j << 4) + lm;
      if (gn >= N) continue;
      float bv = bias ? bias[gn] : 0.0f;
      #pragma unroll
      for (int r = 0; r < 8; ++r) {
        int gm = blockM + wmB + (i << 4) + (half << 3) + r;
        if (gm >= M) continue;
        float v = acc[i][j][r] + bv;
        if (leaky) v = v > 0.0f ? v : LEAKY_SLOPE * v;
        if (outF)  outF[(size_t)gm * N + gn] = v;
        if (outBF) {
          size_t o = bfTrans ? (size_t)gn * ldBF + gm : (size_t)gm * ldBF + gn;
          outBF[o] = f2bf(v);
        }
      }
    }
  }
}

// ------------------------------------------------------------------ row reductions
__global__ void row_reduce(const float* __restrict__ X, int n, int d,
                           float* __restrict__ out, int doSqrt, int doAbs) {
  __shared__ float red[256];
  int row = blockIdx.x;
  const float* Xr = X + (size_t)row * d;
  float s = 0.f;
  for (int c = threadIdx.x; c < d; c += blockDim.x) {
    float v = Xr[c];
    s += doAbs ? fabsf(v) : v * v;
  }
  red[threadIdx.x] = s;
  __syncthreads();
  for (int st = 128; st > 0; st >>= 1) {
    if ((int)threadIdx.x < st) red[threadIdx.x] += red[threadIdx.x + st];
    __syncthreads();
  }
  if (threadIdx.x == 0) out[row] = doSqrt ? sqrtf(red[0]) : red[0];
}

// ------------------------------------------------------------------ gram -> cosine distance
__global__ void dist_kernel(float* __restrict__ g, const float* __restrict__ w, int n) {
  long long total = (long long)n * n;
  long long i = (long long)blockIdx.x * blockDim.x + threadIdx.x;
  long long stride = (long long)gridDim.x * blockDim.x;
  for (; i < total; i += stride) {
    int r = (int)(i / n), c = (int)(i % n);
    float denom = fmaxf(w[r] * w[c], 1e-8f);
    g[i] = 1.0f - g[i] / denom;
  }
}

// ------------------------------------------------------------------ kth-smallest selection
__global__ void zero_i32(int* p, int n) {
  int i = blockIdx.x * blockDim.x + threadIdx.x;
  int stride = gridDim.x * blockDim.x;
  for (; i < n; i += stride) p[i] = 0;
}

__global__ void init_sel(const int* __restrict__ epn, int n,
                         float* __restrict__ range, int* __restrict__ rank) {
  if (blockIdx.x == 0 && threadIdx.x == 0) {
    range[0] = -0.5f;
    range[1] = 2.5f;
    rank[0]  = epn[0] * n;
  }
}

__global__ void hist_kernel(const float* __restrict__ d, long long count,
                            const float* __restrict__ range, int* __restrict__ hist,
                            int clampAll) {
  float lo = range[0], hi = range[1];
  float scale = (float)HBINS / (hi - lo);
  long long i = (long long)blockIdx.x * blockDim.x + threadIdx.x;
  long long stride = (long long)gridDim.x * blockDim.x;
  for (; i < count; i += stride) {
    float v = d[i];
    int b = (int)floorf((v - lo) * scale);
    if (clampAll) {
      b = b < 0 ? 0 : (b > HBINS - 1 ? HBINS - 1 : b);
      atomicAdd(&hist[b], 1);
    } else if (b >= 0 && b < HBINS) {
      atomicAdd(&hist[b], 1);
    }
  }
}

__global__ void select_kernel(const int* __restrict__ hist,
                              const float* __restrict__ rangeIn,
                              float* __restrict__ rangeOut,
                              int* __restrict__ rank, int finalPass,
                              float* __restrict__ thrOut) {
  if (blockIdx.x != 0 || threadIdx.x != 0) return;
  long long k = rank[0];
  float lo = rangeIn[0], hi = rangeIn[1];
  float bw = (hi - lo) / (float)HBINS;
  long long cum = 0;
  int b = 0;
  for (; b < HBINS; ++b) {
    int h = hist[b];
    if (cum + h > k) break;
    cum += h;
  }
  if (b >= HBINS) b = HBINS - 1;
  if (finalPass) {
    thrOut[0] = lo + (float)(b + 1) * bw;
  } else {
    rangeOut[0] = lo + (float)b * bw;
    rangeOut[1] = lo + (float)(b + 1) * bw;
    rank[0] = (int)(k - cum);
  }
}

// ------------------------------------------------------------------ adjacency build + normalize
__global__ void build_adj(const float* __restrict__ dist, const float* __restrict__ thrP,
                          int n, float* __restrict__ adj) {
  float thr = thrP[0];
  long long total = (long long)n * n;
  long long i = (long long)blockIdx.x * blockDim.x + threadIdx.x;
  long long stride = (long long)gridDim.x * blockDim.x;
  for (; i < total; i += stride) {
    int r = (int)(i / n), c = (int)(i % n);
    float dij = dist[i];
    float dji = dist[(size_t)c * n + r];
    float aij = (dij <= thr && r != c) ? (1.0f - dij) : 0.0f;
    float aji = (dji <= thr && r != c) ? (1.0f - dji) : 0.0f;
    adj[i] = fmaxf(aij, aji) + (r == c ? 1.0f : 0.0f);
  }
}

__global__ void norm_rows_bf(const float* __restrict__ A, const float* __restrict__ rs,
                             int n, __bf16* __restrict__ out) {
  long long total = (long long)n * n;
  long long i = (long long)blockIdx.x * blockDim.x + threadIdx.x;
  long long stride = (long long)gridDim.x * blockDim.x;
  for (; i < total; i += stride) {
    int r = (int)(i / n);
    out[i] = f2bf(A[i] / fmaxf(rs[r], 1e-12f));
  }
}

// ------------------------------------------------------------------ host
extern "C" void kernel_launch(void* const* d_in, const int* in_sizes, int n_in,
                              void* d_out, int out_size, void* d_ws, size_t ws_size,
                              hipStream_t stream) {
  const int*   epn  = (const int*)d_in[0];
  const float* x    = (const float*)d_in[1];
  const float* adj0 = (const float*)d_in[2];
  const float* w1 = (const float*)d_in[3];
  const float* b1 = (const float*)d_in[4];
  const float* w2 = (const float*)d_in[5];
  const float* b2 = (const float*)d_in[6];
  const float* w3 = (const float*)d_in[7];
  const float* b3 = (const float*)d_in[8];

  const int n  = (int)llround(sqrt((double)in_sizes[2]));  // 4000
  const int d  = in_sizes[1] / n;                          // 1000
  const int h1 = in_sizes[3] / d;                          // 400
  const int h2 = in_sizes[5] / h1;                         // 400
  const int h3 = in_sizes[7] / h2;                         // 200
  const long long nn = (long long)n * n;

  auto alup = [](int v, int a) { return (v + a - 1) / a * a; };
  const int Kd  = alup(d, 32);        // 1024
  const int Kh1 = alup(h1, 32);       // 416
  const int Kh2 = alup(h2, 32);       // 416
  const int MA  = alup(n, BM);        // 4096 (also covers B-row need alup(n,64)=4032)
  const int NB1 = alup(h1, BN);       // 448
  const int NB2 = alup(h2, BN);
  const int NB3 = alup(h3, BN);

  char* base = (char*)d_ws;
  size_t off = 0;
  auto carve = [&](size_t bytes) -> void* {
    void* p = base + off;
    off = (off + bytes + 255) & ~(size_t)255;
    return p;
  };
  __bf16* xbf   = (__bf16*)carve((size_t)MA * Kd * 2);
  __bf16* w1t   = (__bf16*)carve((size_t)NB1 * Kd * 2);
  __bf16* w2t   = (__bf16*)carve((size_t)NB2 * Kh1 * 2);
  __bf16* w3t   = (__bf16*)carve((size_t)NB3 * Kh2 * 2);
  __bf16* adjbf = (__bf16*)carve((size_t)MA * n * 2);
  __bf16* xwT   = (__bf16*)carve((size_t)NB1 * n * 2);   // holds XW^T, reused per layer
  float*  h1f   = (float*)carve((size_t)n * h1 * 4);
  __bf16* h1bf  = (__bf16*)carve((size_t)MA * Kh1 * 2);
  float*  h2f   = (float*)carve((size_t)n * h2 * 4);
  __bf16* h2bf  = (__bf16*)carve((size_t)MA * Kh2 * 2);
  float*  gram  = (float*)carve((size_t)nn * 4);
  float*  adjW  = (float*)carve((size_t)nn * 4);
  float*  norms = (float*)carve((size_t)n * 4);
  float*  rsum  = (float*)carve((size_t)n * 4);
  int*    hist  = (int*)carve(HBINS * 4);
  float*  rngA  = (float*)carve(2 * 4);
  float*  rngB  = (float*)carve(2 * 4);
  int*    rank  = (int*)carve(4);
  float*  thr   = (float*)carve(4);

  const dim3 T(256);
  const dim3 GS(4096);
  auto gemm = [&](const __bf16* A, int lda, const __bf16* B, int ldb,
                  int M, int N, int K, const float* bias, int leaky,
                  float* oF, __bf16* oBF, int ldBF, int bfTrans) {
    dim3 g((N + BN - 1) / BN, (M + BM - 1) / BM);
    gemm_bf16_wmma<<<g, T, 0, stream>>>(A, lda, B, ldb, M, N, K,
                                        bias, leaky, oF, oBF, ldBF, bfTrans);
  };
  auto gen_adj = [&](const float* Hf, const __bf16* Hbf, int h, int Kh) {
    row_reduce<<<n, T, 0, stream>>>(Hf, n, h, norms, 1, 0);
    gemm(Hbf, Kh, Hbf, Kh, n, n, Kh, nullptr, 0, gram, nullptr, 0, 0);  // H @ H^T
    dist_kernel<<<GS, T, 0, stream>>>(gram, norms, n);
    init_sel<<<1, 1, 0, stream>>>(epn, n, rngA, rank);
    zero_i32<<<16, T, 0, stream>>>(hist, HBINS);
    hist_kernel<<<GS, T, 0, stream>>>(gram, nn, rngA, hist, 1);
    select_kernel<<<1, 1, 0, stream>>>(hist, rngA, rngB, rank, 0, thr);
    zero_i32<<<16, T, 0, stream>>>(hist, HBINS);
    hist_kernel<<<GS, T, 0, stream>>>(gram, nn, rngB, hist, 0);
    select_kernel<<<1, 1, 0, stream>>>(hist, rngB, rngB, rank, 1, thr);
    build_adj<<<GS, T, 0, stream>>>(gram, thr, n, adjW);
    row_reduce<<<n, T, 0, stream>>>(adjW, n, n, rsum, 0, 1);
    norm_rows_bf<<<GS, T, 0, stream>>>(adjW, rsum, n, adjbf);
  };

  // padded bf16 operands (K-pads zero-filled)
  cvt_pad  <<<GS, T, 0, stream>>>(x,    xbf,   n,  d,  MA,  Kd);
  cvt_pad_t<<<GS, T, 0, stream>>>(w1,   w1t,   h1, d,  NB1, Kd);
  cvt_pad_t<<<GS, T, 0, stream>>>(w2,   w2t,   h2, h1, NB2, Kh1);
  cvt_pad_t<<<GS, T, 0, stream>>>(w3,   w3t,   h3, h2, NB3, Kh2);
  cvt_pad  <<<GS, T, 0, stream>>>(adj0, adjbf, n,  n,  MA,  n);

  // layer 1
  gemm(xbf, Kd, w1t, Kd, n, h1, Kd, nullptr, 0, nullptr, xwT, n, 1);      // XW1^T
  gemm(adjbf, n, xwT, n, n, h1, n, b1, 1, h1f, h1bf, Kh1, 0);             // H1
  zero_cols_bf<<<64, T, 0, stream>>>(h1bf, n, h1, Kh1, Kh1);
  gen_adj(h1f, h1bf, h1, Kh1);
  // layer 2
  gemm(h1bf, Kh1, w2t, Kh1, n, h2, Kh1, nullptr, 0, nullptr, xwT, n, 1);  // XW2^T
  gemm(adjbf, n, xwT, n, n, h2, n, b2, 1, h2f, h2bf, Kh2, 0);             // H2
  zero_cols_bf<<<64, T, 0, stream>>>(h2bf, n, h2, Kh2, Kh2);
  gen_adj(h2f, h2bf, h2, Kh2);
  // layer 3
  gemm(h2bf, Kh2, w3t, Kh2, n, h3, Kh2, nullptr, 0, nullptr, xwT, n, 1);  // XW3^T
  gemm(adjbf, n, xwT, n, n, h3, n, b3, 1, (float*)d_out, nullptr, 0, 0);  // output
}
